// GeometricStructureEmbedding_43894565765545
// MI455X (gfx1250) — matrix operand order repaired and checked by reading, not compiled, not run on hardware
//
#include <hip/hip_runtime.h>
#include <hip/hip_bf16.h>
#include <math.h>

typedef __attribute__((ext_vector_type(16))) _Float16 v16h;
typedef __attribute__((ext_vector_type(8)))  float    v8f;

#define NPTS   512
#define DMODEL 256

// ---------------------------------------------------------------------------
// Pack W^T (for out = emb @ W^T) into the WMMA-B (32x16 f16) lane layout:
// flat = ((ntile*8 + kstep)*32 + lane)*16 + h
//   column n = ntile*16 + (lane & 15)
//   k        = kstep*32 + (lane >> 4)*8 + (h & 7) + (h >> 3)*16     (lane<16: K 0..7,16..23, lane>=16: K 8..15,24..31)
// B[k][n] = W[n][k]
// ---------------------------------------------------------------------------
__global__ __launch_bounds__(256) void pack_weights_kernel(const float* __restrict__ W,
                                                           _Float16* __restrict__ Bp) {
  int flat = blockIdx.x * blockDim.x + threadIdx.x;
  if (flat >= DMODEL * DMODEL) return;
  int h     = flat & 15;
  int lane  = (flat >> 4) & 31;
  int kstep = (flat >> 9) & 7;
  int ntile = flat >> 12;
  int n = ntile * 16 + (lane & 15);
  int k = kstep * 32 + (lane >> 4) * 8 + (h & 7) + (h >> 3) * 16;
  Bp[flat] = (_Float16)W[n * DMODEL + k];
}

// ---------------------------------------------------------------------------
// 3-NN per point (excluding self). Strict '<' insertion keeps equal distances
// ordered by lower index (matches lax.top_k tie behavior); slot 0 is self.
// Stores ref_vec[n][j][c] = p[knn_j] - p[n]  (512 x 9 floats).
// ---------------------------------------------------------------------------
__global__ __launch_bounds__(256) void knn_kernel(const float* __restrict__ pts,
                                                  float* __restrict__ refvec) {
  int n = blockIdx.x * blockDim.x + threadIdx.x;
  if (n >= NPTS) return;
  float px = pts[n * 3 + 0], py = pts[n * 3 + 1], pz = pts[n * 3 + 2];
  float bd0 = 3.4e38f, bd1 = 3.4e38f, bd2 = 3.4e38f, bd3 = 3.4e38f;
  int   bi0 = 0, bi1 = 0, bi2 = 0, bi3 = 0;
  for (int m = 0; m < NPTS; ++m) {
    float dx = pts[m * 3 + 0] - px;
    float dy = pts[m * 3 + 1] - py;
    float dz = pts[m * 3 + 2] - pz;
    float d2 = dx * dx + dy * dy + dz * dz;
    if (d2 < bd3) {
      if (d2 < bd2) {
        bd3 = bd2; bi3 = bi2;
        if (d2 < bd1) {
          bd2 = bd1; bi2 = bi1;
          if (d2 < bd0) { bd1 = bd0; bi1 = bi0; bd0 = d2; bi0 = m; }
          else          { bd1 = d2;  bi1 = m; }
        } else { bd2 = d2; bi2 = m; }
      } else { bd3 = d2; bi3 = m; }
    }
  }
  int nb0 = bi1, nb1 = bi2, nb2 = bi3;   // drop self (slot 0)
  int nb[3] = {nb0, nb1, nb2};
  for (int j = 0; j < 3; ++j) {
    int q = nb[j];
    refvec[n * 9 + j * 3 + 0] = pts[q * 3 + 0] - px;
    refvec[n * 9 + j * 3 + 1] = pts[q * 3 + 1] - py;
    refvec[n * 9 + j * 3 + 2] = pts[q * 3 + 2] - pz;
  }
}

// ---------------------------------------------------------------------------
// Fused main kernel.
// Block = (n, 64-row m-chunk); 4 waves, each wave owns a 16-row M-tile.
// Per lane: compute the 4 index scalars, expand them into the WMMA A-operand
// registers (4 embeddings x 8 K-steps of v16h), then sweep 16 N-tiles with
// v_wmma_f32_16x16x32_f16, fusing max-over-k + biases into the f32 store.
// A layout (16-bit A 16x32): lane<16 -> row=lane, halfs = K{0..7,16..23};
// lane>=16 -> row=lane-16, halfs = K{8..15,24..31}. For K pair (2i,2i+1) the
// values are (sin(x*w_i), cos(x*w_i)), w_i = exp(-i*ln(1e4)/128).
// ---------------------------------------------------------------------------
__global__ __launch_bounds__(128, 1) void gse_main_kernel(
    const float* __restrict__ pts,
    const v16h*  __restrict__ Bdp,
    const v16h*  __restrict__ Bap,
    const float* __restrict__ bd,
    const float* __restrict__ ba,
    const float* __restrict__ refvec,
    float*       __restrict__ out)
{
  const int n      = blockIdx.x;
  const int mchunk = blockIdx.y;
  const int wave   = threadIdx.x >> 5;
  const int lane   = threadIdx.x & 31;
  const int rloc   = lane & 15;
  const int s      = lane >> 4;
  const int m      = mchunk * 64 + wave * 16 + rloc;

  // ---- index scalars for this lane's embedding row ----
  float pnx = pts[n * 3 + 0], pny = pts[n * 3 + 1], pnz = pts[n * 3 + 2];
  float ax = pts[m * 3 + 0] - pnx;
  float ay = pts[m * 3 + 1] - pny;
  float az = pts[m * 3 + 2] - pnz;
  float sq = ax * ax + ay * ay + az * az;

  const float INV_SIGMA_D = 5.0f;                      // 1/0.2
  const float FACTOR_A    = 3.81971863420548805845f;   // 180/(15*pi)

  float xv[4];
  xv[0] = sqrtf(sq) * INV_SIGMA_D;
  #pragma unroll
  for (int j = 0; j < 3; ++j) {
    float rx = refvec[n * 9 + j * 3 + 0];
    float ry = refvec[n * 9 + j * 3 + 1];
    float rz = refvec[n * 9 + j * 3 + 2];
    float cx = ry * az - rz * ay;
    float cy = rz * ax - rx * az;
    float cz = rx * ay - ry * ax;
    float sv = sqrtf(cx * cx + cy * cy + cz * cz);
    float cv = rx * ax + ry * ay + rz * az;
    xv[j + 1] = atan2f(sv, cv) * FACTOR_A;
  }

  // ---- build all A operands in registers (4 embeddings x 8 K-steps) ----
  const float C = 0.07195578718471754f;  // ln(10000)/128
  v16h A[4][8];
  #pragma unroll
  for (int t = 0; t < 8; ++t) {
    #pragma unroll
    for (int hf = 0; hf < 2; ++hf) {
      #pragma unroll
      for (int jj = 0; jj < 4; ++jj) {
        int i = 16 * t + 8 * hf + 4 * s + jj;     // frequency index owned by this lane
        float w = __expf(-C * (float)i);
        #pragma unroll
        for (int e = 0; e < 4; ++e) {
          float sn, cs;
          __sincosf(xv[e] * w, &sn, &cs);
          A[e][t][8 * hf + 2 * jj]     = (_Float16)sn;
          A[e][t][8 * hf + 2 * jj + 1] = (_Float16)cs;
        }
      }
    }
  }

  const size_t rowbase =
      ((size_t)n * NPTS + (size_t)(mchunk * 64 + wave * 16)) * DMODEL;

  // ---- sweep the 16 output N-tiles ----
  #pragma unroll 1
  for (int nt = 0; nt < 16; ++nt) {
    v8f accD = {}; v8f acc0 = {}; v8f acc1 = {}; v8f acc2 = {};
    #pragma unroll
    for (int t = 0; t < 8; ++t) {
      v16h bD = Bdp[(nt * 8 + t) * 32 + lane];
      v16h bA = Bap[(nt * 8 + t) * 32 + lane];
      accD = __builtin_amdgcn_wmma_f32_16x16x32_f16(false, A[0][t], false, bD,
                                                    (short)0, accD, false, false);
      acc0 = __builtin_amdgcn_wmma_f32_16x16x32_f16(false, A[1][t], false, bA,
                                                    (short)0, acc0, false, false);
      acc1 = __builtin_amdgcn_wmma_f32_16x16x32_f16(false, A[2][t], false, bA,
                                                    (short)0, acc1, false, false);
      acc2 = __builtin_amdgcn_wmma_f32_16x16x32_f16(false, A[3][t], false, bA,
                                                    (short)0, acc2, false, false);
    }
    int col = nt * 16 + rloc;
    float biasSum = bd[col] + ba[col];
    // D/C layout: VGPR v, lane -> row = v + 8*(lane>=16), col = lane&15
    #pragma unroll
    for (int v = 0; v < 8; ++v) {
      int rowl = v + 8 * s;
      float val = accD[v] + fmaxf(fmaxf(acc0[v], acc1[v]), acc2[v]) + biasSum;
      out[rowbase + (size_t)rowl * DMODEL + col] = val;
    }
  }
}

// ---------------------------------------------------------------------------
extern "C" void kernel_launch(void* const* d_in, const int* in_sizes, int n_in,
                              void* d_out, int out_size, void* d_ws, size_t ws_size,
                              hipStream_t stream) {
  const float* pts = (const float*)d_in[0];
  const float* Wd  = (const float*)d_in[1];
  const float* bd  = (const float*)d_in[2];
  const float* Wa  = (const float*)d_in[3];
  const float* ba  = (const float*)d_in[4];
  float* out = (float*)d_out;

  _Float16* Bdp = (_Float16*)d_ws;                       // 65536 f16
  _Float16* Bap = Bdp + DMODEL * DMODEL;                 // 65536 f16
  float*  refvec = (float*)(Bap + DMODEL * DMODEL);      // 512*9 f32

  pack_weights_kernel<<<(DMODEL * DMODEL) / 256, 256, 0, stream>>>(Wd, Bdp);
  pack_weights_kernel<<<(DMODEL * DMODEL) / 256, 256, 0, stream>>>(Wa, Bap);
  knn_kernel<<<2, 256, 0, stream>>>(pts, refvec);

  dim3 grid(NPTS, 8);
  gse_main_kernel<<<grid, 128, 0, stream>>>(pts, (const v16h*)Bdp, (const v16h*)Bap,
                                            bd, ba, refvec, out);
}
